// DbrxBlock_76879914598688
// MI455X (gfx1250) — matrix operand order repaired and checked by reading, not compiled, not run on hardware
//
#include <hip/hip_runtime.h>

// ---------------- problem constants ----------------
#define BB   2
#define SS   2048
#define DD   1024
#define HH   16
#define KVH_ 4
#define HD_  64
#define HDV  (HD_ + 16)              // V rows + ones/zeros tile for row-sum WMMA
#define EE   8
#define FF   2048
#define NQKV (DD + 2 * KVH_ * HD_)   // 1536
#define BS   (BB * SS)               // 4096 tokens
#define SPAD (SS + 32)               // padded key stride for transposed V
#define CLIP_ 8.0f
#define ROPE_BASE_ 10000.0f
#define EPS_ 1e-5f
#define SM_SCALE_ 0.125f             // 1/sqrt(64)

typedef __attribute__((ext_vector_type(16))) __bf16 v16bf;
typedef __attribute__((ext_vector_type(8)))  __bf16 v8bf;
typedef __attribute__((ext_vector_type(8)))  float  v8f;

__device__ __forceinline__ float wave_red_add(float v) {
  for (int off = 16; off > 0; off >>= 1) v += __shfl_xor(v, off, 32);
  return v;
}

// max-reduce across the 16 lanes of each wave half, LDS-free where possible
#if __has_builtin(__builtin_amdgcn_mov_dpp8)
template <int M>
__device__ __forceinline__ float dpp8_xor_max(float v) {
  constexpr int sel = ((0 ^ M) << 0)  | ((1 ^ M) << 3)  | ((2 ^ M) << 6)  | ((3 ^ M) << 9) |
                      ((4 ^ M) << 12) | ((5 ^ M) << 15) | ((6 ^ M) << 18) | ((7 ^ M) << 21);
  return fmaxf(v, __builtin_bit_cast(float,
      __builtin_amdgcn_mov_dpp8(__builtin_bit_cast(int, v), sel)));
}
#endif

__device__ __forceinline__ float red16_max(float v) {
#if __has_builtin(__builtin_amdgcn_mov_dpp8)
  v = dpp8_xor_max<1>(v);
  v = dpp8_xor_max<2>(v);
  v = dpp8_xor_max<4>(v);
#else
  v = fmaxf(v, __shfl_xor(v, 1, 32));
  v = fmaxf(v, __shfl_xor(v, 2, 32));
  v = fmaxf(v, __shfl_xor(v, 4, 32));
#endif
#if __has_builtin(__builtin_amdgcn_permlane16)
  {
    const int x = __builtin_bit_cast(int, v);
    const int y = __builtin_amdgcn_permlane16(x, x, 0xFEDCBA98, 0x76543210, true, false);
    v = fmaxf(v, __builtin_bit_cast(float, y));
  }
#else
  v = fmaxf(v, __shfl_xor(v, 8, 32));
#endif
  return v;
}

// ---------------- f32 -> bf16 convert (layout preserved) ----------------
__global__ void cvt_bf16_kernel(const float* __restrict__ src,
                                __bf16* __restrict__ dst, long n) {
  long i = (long)blockIdx.x * blockDim.x + threadIdx.x;
  long stride = (long)gridDim.x * blockDim.x;
  for (; i < n; i += stride) dst[i] = (__bf16)src[i];
}

// ---------------- f32 [K][N] -> bf16 [N][K] tiled transpose-convert ----------------
// grid (N/32, K/32, batch); 256 threads; batch stride = K*N.
template <int K, int N>
__global__ void transpose_cvt_kernel(const float* __restrict__ src,
                                     __bf16* __restrict__ dst) {
  __shared__ __bf16 tile[32][33];
  const size_t base = (size_t)blockIdx.z * K * N;
  const int n0 = blockIdx.x * 32, k0 = blockIdx.y * 32;
  const int c = threadIdx.x & 31, r8 = threadIdx.x >> 5;
  #pragma unroll
  for (int r = r8; r < 32; r += 8)
    tile[r][c] = (__bf16)src[base + (size_t)(k0 + r) * N + n0 + c];
  __syncthreads();
  #pragma unroll
  for (int r = r8; r < 32; r += 8)
    dst[base + (size_t)(n0 + r) * K + k0 + c] = tile[c][r];
}

// ---------------- init transposed V: ones row (hd==HD_), zeros elsewhere ----------------
// vt: [BB*KVH_][HDV][SPAD]; RoPE overwrites rows 0..HD_-1, cols 0..SS-1.
__global__ void init_vt_kernel(__bf16* __restrict__ vt) {
  const size_t total = (size_t)BB * KVH_ * HDV * SPAD;
  size_t i = (size_t)blockIdx.x * blockDim.x + threadIdx.x;
  const size_t stride = (size_t)gridDim.x * blockDim.x;
  for (; i < total; i += stride) {
    const int row = (int)((i / SPAD) % HDV);
    vt[i] = (row == HD_) ? (__bf16)1.0f : (__bf16)0.0f;
  }
}

// ---------------- LayerNorm (f32 in, bf16 out) ----------------
__global__ void ln_bf16_kernel(const float* __restrict__ x,
                               const float* __restrict__ w,
                               __bf16* __restrict__ out) {
  const int row = blockIdx.x;
  const int tid = threadIdx.x;
  const float* xr = x + (size_t)row * DD;
  __shared__ float red[8];
  __shared__ float stat[2];

  float s = 0.f;
  for (int i = tid; i < DD; i += 256) s += xr[i];
  s = wave_red_add(s);
  if ((tid & 31) == 0) red[tid >> 5] = s;
  __syncthreads();
  if (tid == 0) { float t = 0.f; for (int i = 0; i < 8; i++) t += red[i]; stat[0] = t / DD; }
  __syncthreads();
  const float mu = stat[0];

  float v = 0.f;
  for (int i = tid; i < DD; i += 256) { float d = xr[i] - mu; v += d * d; }
  v = wave_red_add(v);
  __syncthreads();
  if ((tid & 31) == 0) red[tid >> 5] = v;
  __syncthreads();
  if (tid == 0) { float t = 0.f; for (int i = 0; i < 8; i++) t += red[i]; stat[1] = rsqrtf(t / DD + EPS_); }
  __syncthreads();
  const float rs = stat[1];

  for (int i = tid; i < DD; i += 256)
    out[(size_t)row * DD + i] = (__bf16)((xr[i] - mu) * rs * w[i]);
}

// ---------------- bf16 WMMA GEMM, B always [N][K] row-major ----------------
// C[M,N](f32) = op(clip / rowScale / resid / accum)( A[M,K](bf16) * B^T-layout )
// Block = 256 threads = 8 waves; wave tile 32x64; block tile 128x128.
template <int N, int K>
__global__ __launch_bounds__(256)
void gemm_bf16_wmma_kernel(const __bf16* __restrict__ A,
                           const __bf16* __restrict__ Bm,
                           float* __restrict__ C,
                           const float* __restrict__ resid,
                           const float* __restrict__ rowScale, int rsStride,
                           int accum, float clipv) {
  const int lane = threadIdx.x & 31;
  const int wid  = threadIdx.x >> 5;
  const int hlf  = lane >> 4;
  const int lm   = lane & 15;
  const int m0 = blockIdx.y * 128 + (wid & 3) * 32;
  const int n0 = blockIdx.x * 128 + (wid >> 2) * 64;

  v8f acc[2][4] = {};

  // A-frag per lane: row = lane%16, k in {base..base+7, base+16..base+23}, base = 8*(lane/16)
  const __bf16* aptr0 = A + (size_t)(m0 + lm) * K + hlf * 8;
  const __bf16* aptr1 = aptr0 + (size_t)16 * K;
  // B-frag per lane: col = n0 + t*16 + lane%16, k in {base..base+7, base+16..base+23}, base = 8*(lane/16)
  const __bf16* bptr  = Bm + (size_t)(n0 + lm) * K + hlf * 8;

  union AF { v16bf v; v8bf h[2]; };

  for (int k0 = 0; k0 < K; k0 += 32) {
    AF a0, a1;
    a0.h[0] = *(const v8bf*)(aptr0 + k0);
    a0.h[1] = *(const v8bf*)(aptr0 + k0 + 16);
    a1.h[0] = *(const v8bf*)(aptr1 + k0);
    a1.h[1] = *(const v8bf*)(aptr1 + k0 + 16);
    __builtin_prefetch((const void*)(aptr0 + k0 + 32), 0, 1);   // speculative; OOB dropped

    #pragma unroll
    for (int t = 0; t < 4; t++) {
      AF b;
      const __bf16* bp = bptr + (size_t)(t * 16) * K + k0;
      b.h[0] = *(const v8bf*)(bp);
      b.h[1] = *(const v8bf*)(bp + 16);
      acc[0][t] = __builtin_amdgcn_wmma_f32_16x16x32_bf16(false, a0.v, false, b.v,
                                                          (short)0, acc[0][t], false, false);
      acc[1][t] = __builtin_amdgcn_wmma_f32_16x16x32_bf16(false, a1.v, false, b.v,
                                                          (short)0, acc[1][t], false, false);
    }
  }

  #pragma unroll
  for (int g = 0; g < 2; g++) {
    #pragma unroll
    for (int t = 0; t < 4; t++) {
      const int col = n0 + t * 16 + lm;
      #pragma unroll
      for (int i = 0; i < 8; i++) {
        const int row = m0 + g * 16 + i + hlf * 8;   // C-frag: vgpr i -> row i (+8 upper lanes)
        float v = acc[g][t][i];
        if (clipv > 0.f) v = fminf(fmaxf(v, -clipv), clipv);
        if (rowScale) v *= rowScale[(size_t)row * rsStride];
        const size_t idx = (size_t)row * N + col;
        if (resid) v += resid[idx];
        if (accum) v += C[idx];
        C[idx] = v;
      }
    }
  }
}

// ---------------- RoPE + split + bf16 convert ----------------
// qkv f32 [BS, NQKV] -> qb [BS, H*HD] (bf16), kb [BS+pad, KVH*HD] (bf16),
// vt [B][KVH][HDV][SPAD] (bf16, key-contiguous = transposed for PV B-frags)
__global__ void rope_split_kernel(const float* __restrict__ qkv,
                                  __bf16* __restrict__ qb,
                                  __bf16* __restrict__ kb,
                                  __bf16* __restrict__ vt) {
  const int t = blockIdx.x;           // token
  const int pos = t & (SS - 1);       // sequence position
  const int bb  = t >> 11;            // batch (SS = 2048)
  const float* src = qkv + (size_t)t * NQKV;

  for (int i = threadIdx.x; i < HH * 32; i += 256) {   // Q rotate
    const int h = i >> 5, j = i & 31;
    const float x1 = src[h * HD_ + j], x2 = src[h * HD_ + j + 32];
    const float ang = pos * __powf(ROPE_BASE_, -(float)j * (1.0f / 32.0f));
    const float c = __cosf(ang), sn = __sinf(ang);
    qb[(size_t)t * DD + h * HD_ + j]      = (__bf16)(x1 * c - x2 * sn);
    qb[(size_t)t * DD + h * HD_ + j + 32] = (__bf16)(x1 * sn + x2 * c);
  }
  for (int i = threadIdx.x; i < KVH_ * 32; i += 256) { // K rotate
    const int h = i >> 5, j = i & 31;
    const float* ks = src + HH * HD_;
    const float x1 = ks[h * HD_ + j], x2 = ks[h * HD_ + j + 32];
    const float ang = pos * __powf(ROPE_BASE_, -(float)j * (1.0f / 32.0f));
    const float c = __cosf(ang), sn = __sinf(ang);
    kb[(size_t)t * (KVH_ * HD_) + h * HD_ + j]      = (__bf16)(x1 * c - x2 * sn);
    kb[(size_t)t * (KVH_ * HD_) + h * HD_ + j + 32] = (__bf16)(x1 * sn + x2 * c);
  }
  for (int i = threadIdx.x; i < KVH_ * HD_; i += 256) { // V transpose store
    const int kvh = i >> 6, hd = i & 63;
    vt[(((size_t)bb * KVH_ + kvh) * HDV + hd) * SPAD + pos] =
        (__bf16)src[HH * HD_ + KVH_ * HD_ + i];
  }
}

// ---------------- flash attention (causal, GQA) ----------------
// grid (S/16, H, B), one wave per block. 16-query tile, 32 keys / iter.
// kb padded by >=32 key rows; vt rows padded to SPAD and pre-zeroed.
// 5th V tile has a ones column: o[4] accumulates the rescaled row-sum (li)
// entirely inside the WMMA pipeline (no per-iteration sum reduction).
// Single-wave workgroup: LDS relay ordering is guaranteed by in-order DS ops
// within a wave; wave_barrier() is a zero-instruction compiler fence.
__global__ void flash_attn_wmma_kernel(const __bf16* __restrict__ qb,
                                       const __bf16* __restrict__ kb,
                                       const __bf16* __restrict__ vt,
                                       __bf16* __restrict__ ob) {
  const int qt  = blockIdx.x;
  const int h   = blockIdx.y;
  const int b   = blockIdx.z;
  const int kvh = h >> 2;                        // H / KVH = 4
  const int lane = threadIdx.x & 31;
  const int hlf  = lane >> 4;
  const int lm   = lane & 15;

  __shared__ __bf16 pl[16][32];                  // P tile (C-layout -> A-layout relay)

  union AF { v16bf v; v8bf h2[2]; };

  AF aq[2];                                      // Q: rows 16 x hd 64 (two k=32 chunks)
  {
    const __bf16* qp = qb + (((size_t)(b * SS + qt * 16 + lm)) * HH + h) * HD_;
    #pragma unroll
    for (int u = 0; u < 2; u++) {
      aq[u].h2[0] = *(const v8bf*)(qp + u * 32 + hlf * 8);
      aq[u].h2[1] = *(const v8bf*)(qp + u * 32 + hlf * 8 + 16);
    }
  }

  const __bf16* kbase = kb + ((size_t)(b * SS) * KVH_ + kvh) * HD_;
  const __bf16* vbase = vt + ((size_t)b * KVH_ + kvh) * HDV * SPAD;

  v8f o[5] = {};                                 // 4 output tiles + row-sum tile
  float mi[8];
  #pragma unroll
  for (int i = 0; i < 8; i++) mi[i] = -1e30f;

  const int kend = qt * 16 + 16;                 // causal horizon
  for (int k0 = 0; k0 < kend; k0 += 32) {
    const bool needMask = (k0 + 31 > qt * 16);   // uniform across the wave

    // ---- scores: hoist all K fragments, then 4 WMMAs ----
    AF bk[2][2];
    #pragma unroll
    for (int nt = 0; nt < 2; nt++) {
      const __bf16* kp = kbase + (size_t)(k0 + nt * 16 + lm) * (KVH_ * HD_) + hlf * 8;
      bk[nt][0].h2[0] = *(const v8bf*)(kp);
      bk[nt][0].h2[1] = *(const v8bf*)(kp + 16);
      bk[nt][1].h2[0] = *(const v8bf*)(kp + 32);
      bk[nt][1].h2[1] = *(const v8bf*)(kp + 48);
    }
    float ps[2][8];
    #pragma unroll
    for (int nt = 0; nt < 2; nt++) {
      v8f s = {};
      s = __builtin_amdgcn_wmma_f32_16x16x32_bf16(false, aq[0].v, false, bk[nt][0].v, (short)0, s, false, false);
      s = __builtin_amdgcn_wmma_f32_16x16x32_bf16(false, aq[1].v, false, bk[nt][1].v, (short)0, s, false, false);
      if (needMask) {
        #pragma unroll
        for (int i = 0; i < 8; i++) {
          const int row = qt * 16 + i + hlf * 8;
          const int ck  = k0 + nt * 16 + lm;
          ps[nt][i] = (ck > row) ? -1e30f : s[i] * SM_SCALE_;
        }
      } else {
        #pragma unroll
        for (int i = 0; i < 8; i++) ps[nt][i] = s[i] * SM_SCALE_;
      }
    }

    // ---- online max + rescale (DPP reduction, no LDS) ----
    #pragma unroll
    for (int i = 0; i < 8; i++) {
      const float mv   = red16_max(fmaxf(ps[0][i], ps[1][i]));
      const float mnew = fmaxf(mi[i], mv);
      const float alpha = __expf(mi[i] - mnew);
      const float p0 = __expf(ps[0][i] - mnew);
      const float p1 = __expf(ps[1][i] - mnew);
      mi[i] = mnew;
      #pragma unroll
      for (int t = 0; t < 5; t++) o[t][i] *= alpha;
      pl[i + hlf * 8][lm]      = (__bf16)p0;     // stash P in LDS (row-major tile)
      pl[i + hlf * 8][lm + 16] = (__bf16)p1;
    }
    __builtin_amdgcn_wave_barrier();             // intra-wave DS ordering fence (no instr)

    AF pa;                                       // P as A-fragment: two b128 DS loads
    pa.h2[0] = *(const v8bf*)&pl[lm][hlf * 8];
    pa.h2[1] = *(const v8bf*)&pl[lm][hlf * 8 + 16];

    // ---- PV: hoist all 5 V fragments (keys contiguous in vt), then 5 WMMAs ----
    AF bv[5];
    #pragma unroll
    for (int t = 0; t < 5; t++) {
      const __bf16* vp = vbase + (size_t)(t * 16 + lm) * SPAD + k0 + hlf * 8;
      bv[t].h2[0] = *(const v8bf*)(vp);
      bv[t].h2[1] = *(const v8bf*)(vp + 16);
    }
    #pragma unroll
    for (int t = 0; t < 5; t++)
      o[t] = __builtin_amdgcn_wmma_f32_16x16x32_bf16(false, pa.v, false, bv[t].v,
                                                     (short)0, o[t], false, false);
    __builtin_amdgcn_wave_barrier();             // WAR fence before next P store
  }

  // row-sum li lives in column 0 of tile 4 (lane lm==0 of each half)
  #pragma unroll
  for (int t = 0; t < 4; t++) {
    #pragma unroll
    for (int i = 0; i < 8; i++) {
      const float li  = __shfl(o[4][i], hlf * 16, 32);
      const int   row = qt * 16 + i + hlf * 8;
      ob[(((size_t)(b * SS + row)) * HH + h) * HD_ + t * 16 + lm] = (__bf16)(o[t][i] / li);
    }
  }
}

// ---------------- router: softmax + top-2 + renorm ----------------
__global__ void router_topk_kernel(const __bf16* __restrict__ x2b,
                                   const float* __restrict__ rw,
                                   float* __restrict__ gates) {
  const int token = blockIdx.x * 8 + (threadIdx.x >> 5);
  const int lane  = threadIdx.x & 31;
  float acc[EE];
  #pragma unroll
  for (int e = 0; e < EE; e++) acc[e] = 0.f;
  const __bf16* xr = x2b + (size_t)token * DD;
  for (int d = lane; d < DD; d += 32) {
    const float xv = (float)xr[d];
    #pragma unroll
    for (int e = 0; e < EE; e++) acc[e] += xv * rw[(size_t)d * EE + e];
  }
  #pragma unroll
  for (int e = 0; e < EE; e++)
    for (int off = 16; off > 0; off >>= 1) acc[e] += __shfl_xor(acc[e], off, 32);

  float mx = acc[0];
  #pragma unroll
  for (int e = 1; e < EE; e++) mx = fmaxf(mx, acc[e]);
  float w[EE], sum = 0.f;
  #pragma unroll
  for (int e = 0; e < EE; e++) { w[e] = __expf(acc[e] - mx); sum += w[e]; }
  #pragma unroll
  for (int e = 0; e < EE; e++) w[e] /= sum;
  int e1 = 0;
  #pragma unroll
  for (int e = 1; e < EE; e++) if (w[e] > w[e1]) e1 = e;
  int e2 = (e1 == 0) ? 1 : 0;
  #pragma unroll
  for (int e = 0; e < EE; e++) if (e != e1 && w[e] > w[e2]) e2 = e;
  const float norm = w[e1] + w[e2];
  if (lane < EE) {
    const float g = (lane == e1) ? w[e1] / norm : (lane == e2 ? w[e2] / norm : 0.f);
    gates[(size_t)token * EE + lane] = g;
  }
}

// ---------------- SiLU(h1) * h2 -> bf16 ----------------
__global__ void silu_mul_kernel(const float* __restrict__ h1,
                                const float* __restrict__ h2,
                                __bf16* __restrict__ g, long n) {
  long i = (long)blockIdx.x * blockDim.x + threadIdx.x;
  long stride = (long)gridDim.x * blockDim.x;
  for (; i < n; i += stride) {
    const float a = h1[i];
    g[i] = (__bf16)((a / (1.f + __expf(-a))) * h2[i]);
  }
}

// ---------------- host orchestration ----------------
extern "C" void kernel_launch(void* const* d_in, const int* in_sizes, int n_in,
                              void* d_out, int out_size, void* d_ws, size_t ws_size,
                              hipStream_t stream) {
  const float* hidden  = (const float*)d_in[0];
  const float* ln1w    = (const float*)d_in[1];
  const float* ln2w    = (const float*)d_in[2];
  const float* wqkv    = (const float*)d_in[3];
  const float* outw    = (const float*)d_in[4];
  const float* routerw = (const float*)d_in[5];
  const float* w1      = (const float*)d_in[6];
  const float* v1      = (const float*)d_in[7];
  const float* w2      = (const float*)d_in[8];
  float* out = (float*)d_out;

  char* p = (char*)d_ws;
  auto carve = [&](size_t bytes) -> void* {
    void* r = (void*)p;
    p += (bytes + 255) & ~(size_t)255;
    return r;
  };
  __bf16* xb    = (__bf16*)carve((size_t)BS * DD * 2);
  __bf16* x2b   = (__bf16*)carve((size_t)BS * DD * 2);
  __bf16* wqkvb = (__bf16*)carve((size_t)DD * NQKV * 2);    // [NQKV][D]
  __bf16* outwb = (__bf16*)carve((size_t)DD * DD * 2);      // [D][D] transposed
  __bf16* w1b   = (__bf16*)carve((size_t)EE * FF * DD * 2); // already [F][D]
  __bf16* v1b   = (__bf16*)carve((size_t)EE * FF * DD * 2); // already [F][D]
  __bf16* w2b   = (__bf16*)carve((size_t)EE * FF * DD * 2); // [D][F] per expert
  float*  qkv   = (float*) carve((size_t)BS * NQKV * 4);
  __bf16* qbuf  = (__bf16*)carve((size_t)BS * DD * 2);
  __bf16* kbuf  = (__bf16*)carve((size_t)(BS + 32) * KVH_ * HD_ * 2);      // padded
  __bf16* vtb   = (__bf16*)carve((size_t)BB * KVH_ * HDV * SPAD * 2);     // transposed
  __bf16* attnb = (__bf16*)carve((size_t)BS * DD * 2);
  float*  h1    = (float*) carve((size_t)BS * FF * 4);
  float*  h2    = (float*) carve((size_t)BS * FF * 4);
  __bf16* gb    = (__bf16*)carve((size_t)BS * FF * 2);
  float*  gates = (float*) carve((size_t)BS * EE * 4);

  // weight prep: f32 -> bf16, K-major weights transposed to [N][K]
  { dim3 g(NQKV / 32, DD / 32, 1);
    transpose_cvt_kernel<DD, NQKV><<<g, 256, 0, stream>>>(wqkv, wqkvb); }
  { dim3 g(DD / 32, DD / 32, 1);
    transpose_cvt_kernel<DD, DD><<<g, 256, 0, stream>>>(outw, outwb); }
  { dim3 g(DD / 32, FF / 32, EE);
    transpose_cvt_kernel<FF, DD><<<g, 256, 0, stream>>>(w2, w2b); }
  cvt_bf16_kernel<<<2048, 256, 0, stream>>>(w1, w1b, (long)EE * FF * DD);
  cvt_bf16_kernel<<<2048, 256, 0, stream>>>(v1, v1b, (long)EE * FF * DD);

  // transposed V: ones/zeros init (RoPE fills real rows after)
  init_vt_kernel<<<1024, 256, 0, stream>>>(vtb);

  // LN1 -> xb (bf16)
  ln_bf16_kernel<<<BS, 256, 0, stream>>>(hidden, ln1w, xb);

  // QKV projection with clip: qkv = clip(xb @ wqkv)
  {
    dim3 g(NQKV / 128, BS / 128);
    gemm_bf16_wmma_kernel<NQKV, DD><<<g, 256, 0, stream>>>(
        xb, wqkvb, qkv, nullptr, nullptr, 0, 0, CLIP_);
  }

  // RoPE + head split (bf16, V transposed)
  rope_split_kernel<<<BS, 256, 0, stream>>>(qkv, qbuf, kbuf, vtb);

  // causal GQA flash attention
  {
    dim3 g(SS / 16, HH, BB);
    flash_attn_wmma_kernel<<<g, 32, 0, stream>>>(qbuf, kbuf, vtb, attnb);
  }

  // out projection + residual: out = hidden + attn @ out_w
  {
    dim3 g(DD / 128, BS / 128);
    gemm_bf16_wmma_kernel<DD, DD><<<g, 256, 0, stream>>>(
        attnb, outwb, out, hidden, nullptr, 0, 0, 0.f);
  }

  // LN2 -> x2b, router gates
  ln_bf16_kernel<<<BS, 256, 0, stream>>>(out, ln2w, x2b);
  router_topk_kernel<<<BS / 8, 256, 0, stream>>>(x2b, routerw, gates);

  // MoE: out += gate_e * ((silu(x2 @ w1_e^T) * (x2 @ v1_e^T)) @ w2_e)
  for (int e = 0; e < EE; e++) {
    dim3 gu(FF / 128, BS / 128);
    gemm_bf16_wmma_kernel<FF, DD><<<gu, 256, 0, stream>>>(
        x2b, w1b + (size_t)e * FF * DD, h1, nullptr, nullptr, 0, 0, 0.f);
    gemm_bf16_wmma_kernel<FF, DD><<<gu, 256, 0, stream>>>(
        x2b, v1b + (size_t)e * FF * DD, h2, nullptr, nullptr, 0, 0, 0.f);
    silu_mul_kernel<<<2048, 256, 0, stream>>>(h1, h2, gb, (long)BS * FF);
    dim3 gd(DD / 128, BS / 128);
    gemm_bf16_wmma_kernel<DD, FF><<<gd, 256, 0, stream>>>(
        gb, w2b + (size_t)e * FF * DD, out, nullptr, gates + e, EE, 1, 0.f);
  }
}
